// MultiHeadAttention_65481071397869
// MI455X (gfx1250) — matrix operand order repaired and checked by reading, not compiled, not run on hardware
//
#include <hip/hip_runtime.h>
#include <stdint.h>

typedef unsigned short u16;
typedef __attribute__((ext_vector_type(16))) __bf16 v16bf;
typedef __attribute__((ext_vector_type(8)))  float  v8f;
typedef __attribute__((ext_vector_type(4)))  unsigned u32x4;
typedef __attribute__((ext_vector_type(8)))  int      i32x8;
typedef __attribute__((ext_vector_type(4)))  int      i32x4;

struct alignas(16) U4 { unsigned x, y, z, w; };

union FragU {
    U4    u4[2];
    v16bf v;
};

__device__ __forceinline__ u16 f2bf(float f) {
    union { float f; unsigned u; } c;
    c.f = f;
    unsigned u = c.u;
    u += 0x7FFFu + ((u >> 16) & 1u);   // round-to-nearest-even
    return (u16)(u >> 16);
}

#define WMMA_BF16(a, b, c) \
    __builtin_amdgcn_wmma_f32_16x16x32_bf16(false, (a), false, (b), (short)0, (c), false, false)

// ---------------------------------------------------------------------------
// CDNA5 Tensor Data Mover support (guarded; falls back to manual staging)
// ---------------------------------------------------------------------------
#if __has_builtin(__builtin_amdgcn_tensor_load_to_lds)
#define MHA_USE_TDM 1
#else
#define MHA_USE_TDM 0
#endif

#if __has_builtin(__builtin_amdgcn_s_wait_tensorcnt)
#define MHA_WAIT_TENSOR0() __builtin_amdgcn_s_wait_tensorcnt(0)
#else
#define MHA_WAIT_TENSOR0() asm volatile("s_wait_tensorcnt 0x0" ::: "memory")
#endif

typedef __attribute__((address_space(3))) char lds_char;
__device__ __forceinline__ unsigned mha_lds_off(const void* p) {
    return (unsigned)(unsigned long long)(lds_char*)p;
}

#if MHA_USE_TDM
// 2D tile DMA: global (row-major, stride_elems between rows, 2-byte elements)
// -> LDS contiguous tile_d1 rows of tile_d0 elements.  D# per ISA ch.8.
__device__ __forceinline__ void tdm_load_2d(const void* gptr, unsigned lds_byte,
                                            unsigned tile_d0, unsigned tile_d1,
                                            unsigned stride_elems,
                                            unsigned tensor_d0, unsigned tensor_d1) {
    unsigned long long ga = (unsigned long long)gptr;
    u32x4 g0;
    g0.x = 0x1u;                                  // count=1, user descriptor
    g0.y = lds_byte;                              // lds_addr [63:32]
    g0.z = (unsigned)(ga & 0xFFFFFFFFu);          // global_addr [95:64]
    g0.w = (unsigned)((ga >> 32) & 0x01FFFFFFu)   // global_addr[56:32]
         | (2u << 30);                            // type=2 ("image")
    i32x8 g1;
    g1[0] = (int)(1u << 16);                      // workgroup_mask=0, data_size=1 (2B)
    g1[1] = (int)((tensor_d0 & 0xFFFFu) << 16);   // tensor_dim0[15:0] @ bit48
    g1[2] = (int)(((tensor_d0 >> 16) & 0xFFFFu) | ((tensor_d1 & 0xFFFFu) << 16));
    g1[3] = (int)(((tensor_d1 >> 16) & 0xFFFFu) | ((tile_d0 & 0xFFFFu) << 16));
    g1[4] = (int)(tile_d1 & 0xFFFFu);             // tile_dim1; tile_dim2=0
    g1[5] = (int)stride_elems;                    // tensor_dim0_stride[31:0]
    g1[6] = 0;                                    // stride[47:32], dim1_stride[15:0]
    g1[7] = 0;
    i32x4 z4 = {0, 0, 0, 0};
#if __has_include(<hip/amd_detail/amd_gfx1250_TDM.h>)
    i32x8 z8 = {0, 0, 0, 0, 0, 0, 0, 0};
    __builtin_amdgcn_tensor_load_to_lds(g0, g1, z4, z4, z8, 0);
#else
    __builtin_amdgcn_tensor_load_to_lds(g0, g1, z4, z4, 0);
#endif
}
#endif  // MHA_USE_TDM

// ---------------------------------------------------------------------------
// fp32 -> bf16 conversion
// ---------------------------------------------------------------------------
__global__ void mha_cvt_bf16(const float* __restrict__ src, u16* __restrict__ dst, int n) {
    int i = blockIdx.x * blockDim.x + threadIdx.x;
    if (i < n) dst[i] = f2bf(src[i]);
}

// ---------------------------------------------------------------------------
// GEMM: C[m,n] = sum_k A[m,k] * W[n,k]   (A: MxK bf16, W: NxK bf16, row-major)
// Block tile 128x128, 8 waves (4x2), wave tile 32x64, K-step 64,
// double-buffered LDS staged by the Tensor Data Mover (fallback: manual).
// ---------------------------------------------------------------------------
template <bool OUT_F32>
__global__ __launch_bounds__(256)
void mha_gemm_wmma(const u16* __restrict__ A, const u16* __restrict__ W,
                   void* __restrict__ Cout, int M, int N, int K) {
    __shared__ alignas(16) u16 As[2][128][64];
    __shared__ alignas(16) u16 Bs[2][128][64];

    const int tid   = threadIdx.x;
    const int wave  = tid >> 5;
    const int lane  = tid & 31;
    const int l15   = lane & 15;
    const int lhalf = lane >> 4;
    const int wm    = wave & 3;    // M offset 32*wm
    const int wn    = wave >> 2;   // N offset 64*wn
    const int bm    = blockIdx.x * 128;
    const int bn    = blockIdx.y * 128;

    v8f acc[2][4];
#pragma unroll
    for (int i = 0; i < 2; ++i)
#pragma unroll
        for (int j = 0; j < 4; ++j)
#pragma unroll
            for (int r = 0; r < 8; ++r) acc[i][j][r] = 0.f;

#if MHA_USE_TDM
    auto stage = [&](int buf, int k0) {
        if (tid < 32) {
            tdm_load_2d(A + (size_t)bm * K + k0, mha_lds_off(&As[buf][0][0]),
                        64, 128, (unsigned)K, (unsigned)(K - k0), (unsigned)(M - bm));
            tdm_load_2d(W + (size_t)bn * K + k0, mha_lds_off(&Bs[buf][0][0]),
                        64, 128, (unsigned)K, (unsigned)(K - k0), (unsigned)(N - bn));
        }
    };
#else
    // manual staging: 128x64 bf16 = 1024 x 16B chunks per matrix, 4 per thread
    auto stage = [&](int buf, int k0) {
#pragma unroll
        for (int i = 0; i < 4; ++i) {
            int c = tid + i * 256;
            int r = c >> 3, col = (c & 7) * 8;
            *(U4*)&As[buf][r][col] = *(const U4*)(A + (size_t)(bm + r) * K + k0 + col);
            *(U4*)&Bs[buf][r][col] = *(const U4*)(W + (size_t)(bn + r) * K + k0 + col);
        }
    };
#endif

    stage(0, 0);
#if MHA_USE_TDM
    if (tid < 32) MHA_WAIT_TENSOR0();
#endif
    __syncthreads();

    const int nk = K / 64;
    for (int kt = 0; kt < nk; ++kt) {
        const int cur = kt & 1;
        if (kt + 1 < nk) stage(cur ^ 1, (kt + 1) * 64);

#pragma unroll
        for (int ks = 0; ks < 2; ++ks) {
            // A fragments: lane = M row; 16 contiguous K at lhalf*16
            v16bf af[2];
#pragma unroll
            for (int mi = 0; mi < 2; ++mi) {
                FragU f;
                const u16* p = &As[cur][wm * 32 + mi * 16 + l15][ks * 32];
                f.u4[0] = *(const U4*)(p + lhalf * 16);
                f.u4[1] = *(const U4*)(p + lhalf * 16 + 8);
                af[mi] = f.v;
            }
            // B fragments: lane = N col (= W row); 16 contiguous K at lhalf*16
            v16bf bfr[4];
#pragma unroll
            for (int j = 0; j < 4; ++j) {
                FragU f;
                const u16* p = &Bs[cur][wn * 64 + j * 16 + l15][ks * 32 + lhalf * 16];
                f.u4[0] = ((const U4*)p)[0];
                f.u4[1] = ((const U4*)p)[1];
                bfr[j] = f.v;
            }
#pragma unroll
            for (int mi = 0; mi < 2; ++mi)
#pragma unroll
                for (int j = 0; j < 4; ++j)
                    acc[mi][j] = WMMA_BF16(af[mi], bfr[j], acc[mi][j]);
        }

#if MHA_USE_TDM
        if (tid < 32) MHA_WAIT_TENSOR0();
#endif
        __syncthreads();
    }

    // epilogue: C layout — VGPR r, lane half: M = r + 8*lhalf, N = l15
#pragma unroll
    for (int mi = 0; mi < 2; ++mi)
#pragma unroll
        for (int j = 0; j < 4; ++j) {
            int col = bn + wn * 64 + j * 16 + l15;
#pragma unroll
            for (int r = 0; r < 8; ++r) {
                int row = bm + wm * 32 + mi * 16 + lhalf * 8 + r;
                float v = acc[mi][j][r];
                if (OUT_F32)
                    ((float*)Cout)[(size_t)row * N + col] = v;
                else
                    ((u16*)Cout)[(size_t)row * N + col] = f2bf(v);
            }
        }
}

// ---------------------------------------------------------------------------
// Flash attention: Q,K,V as [B*S, E] bf16, head h in cols h*64..h*64+63.
// grid.x = B*H, grid.y = S/128.  256 threads = 8 waves; wave owns 16 q rows.
// Q/K tiles staged by TDM; V staged manually (needs transpose).
// ---------------------------------------------------------------------------
__global__ __launch_bounds__(256)
void mha_flash_attn(const u16* __restrict__ Qm, const u16* __restrict__ Km,
                    const u16* __restrict__ Vm, u16* __restrict__ Om) {
    constexpr int S = 2048, E = 1024, D = 64, H = 16;
    const int bh = blockIdx.x;
    const int b  = bh / H, h = bh % H;
    const int q0 = blockIdx.y * 128;
    const size_t rowbase = (size_t)b * S;
    const int cb = h * D;

    __shared__ alignas(16) u16 Qs[128][64];
    __shared__ alignas(16) u16 Ks[64][64];
    __shared__ alignas(16) u16 Vts[64][64];       // [d][kv]
    __shared__ alignas(16) u16 Ps[8][16][64];     // per-wave P patch

    const int tid = threadIdx.x, wave = tid >> 5, lane = tid & 31;
    const int l15 = lane & 15, lhalf = lane >> 4;

    // stage Q tile 128x64
#if MHA_USE_TDM
    if (tid < 32) {
        tdm_load_2d(Qm + (rowbase + q0) * E + cb, mha_lds_off(&Qs[0][0]),
                    64, 128, E, (unsigned)(E - cb), (unsigned)(S - q0));
        MHA_WAIT_TENSOR0();
    }
#else
#pragma unroll
    for (int i = 0; i < 4; ++i) {
        int c = tid + i * 256;
        int r = c >> 3, col = (c & 7) * 8;
        *(U4*)&Qs[r][col] = *(const U4*)(Qm + (rowbase + q0 + r) * E + cb + col);
    }
#endif
    __syncthreads();

    // preload Q A-fragments (2 k-steps over D=64), kept in registers
    v16bf aq[2];
#pragma unroll
    for (int ks = 0; ks < 2; ++ks) {
        FragU f;
        const u16* p = &Qs[wave * 16 + l15][ks * 32];
        f.u4[0] = *(const U4*)(p + lhalf * 16);
        f.u4[1] = *(const U4*)(p + lhalf * 16 + 8);
        aq[ks] = f.v;
    }

    float mrow[8], lrow[8];
    v8f o[4];
#pragma unroll
    for (int r = 0; r < 8; ++r) { mrow[r] = -1e30f; lrow[r] = 0.f; }
#pragma unroll
    for (int j = 0; j < 4; ++j)
#pragma unroll
        for (int r = 0; r < 8; ++r) o[j][r] = 0.f;

    const float scale = 0.125f;  // 1/sqrt(64)

    for (int kv0 = 0; kv0 < S; kv0 += 64) {
        __syncthreads();
        // K tile [kv][d] via TDM (fallback manual)
#if MHA_USE_TDM
        if (tid < 32)
            tdm_load_2d(Km + (rowbase + kv0) * E + cb, mha_lds_off(&Ks[0][0]),
                        64, 64, E, (unsigned)(E - cb), 64u);
#else
#pragma unroll
        for (int i = 0; i < 2; ++i) {
            int c = tid + i * 256;
            int r = c >> 3, col = (c & 7) * 8;
            *(U4*)&Ks[r][col] = *(const U4*)(Km + (rowbase + kv0 + r) * E + cb + col);
        }
#endif
        // V tile transposed [d][kv] (manual — TDM cannot transpose)
#pragma unroll
        for (int i = 0; i < 2; ++i) {
            int c = tid + i * 256;
            int r = c >> 3, col = (c & 7) * 8;
            U4 vx = *(const U4*)(Vm + (rowbase + kv0 + r) * E + cb + col);
            const u16* e = (const u16*)&vx;
#pragma unroll
            for (int j = 0; j < 8; ++j) Vts[col + j][r] = e[j];
        }
#if MHA_USE_TDM
        if (tid < 32) MHA_WAIT_TENSOR0();
#endif
        __syncthreads();

        // scores = Q x K^T : per wave 16x64
        v8f sc[4];
#pragma unroll
        for (int j = 0; j < 4; ++j)
#pragma unroll
            for (int r = 0; r < 8; ++r) sc[j][r] = 0.f;
#pragma unroll
        for (int ks = 0; ks < 2; ++ks)
#pragma unroll
            for (int j = 0; j < 4; ++j) {
                FragU f;
                const u16* p = &Ks[j * 16 + l15][ks * 32 + lhalf * 16];
                f.u4[0] = ((const U4*)p)[0];
                f.u4[1] = ((const U4*)p)[1];
                sc[j] = WMMA_BF16(aq[ks], f.v, sc[j]);
            }

        // online softmax (rows live in 16-lane halves per C layout)
#pragma unroll
        for (int r = 0; r < 8; ++r) {
            float mx = -1e30f;
#pragma unroll
            for (int j = 0; j < 4; ++j) mx = fmaxf(mx, sc[j][r]);
            for (int msk = 1; msk < 16; msk <<= 1)
                mx = fmaxf(mx, __shfl_xor(mx, msk, 32));
            mx *= scale;
            float mn = fmaxf(mrow[r], mx);
            float cr = __expf(mrow[r] - mn);
            float rs = 0.f;
#pragma unroll
            for (int j = 0; j < 4; ++j) {
                float pv = __expf(sc[j][r] * scale - mn);
                rs += pv;
                Ps[wave][lhalf * 8 + r][j * 16 + l15] = f2bf(pv);
            }
            for (int msk = 1; msk < 16; msk <<= 1)
                rs += __shfl_xor(rs, msk, 32);
            lrow[r] = lrow[r] * cr + rs;
            mrow[r] = mn;
#pragma unroll
            for (int j = 0; j < 4; ++j) o[j][r] *= cr;
        }

        // O += P x V  (P re-read from LDS in A layout; V from transposed tile)
#pragma unroll
        for (int ks = 0; ks < 2; ++ks) {
            FragU fa;
            const u16* p = &Ps[wave][l15][ks * 32];
            fa.u4[0] = *(const U4*)(p + lhalf * 16);
            fa.u4[1] = *(const U4*)(p + lhalf * 16 + 8);
#pragma unroll
            for (int j = 0; j < 4; ++j) {
                FragU fb;
                const u16* q = &Vts[j * 16 + l15][ks * 32 + lhalf * 16];
                fb.u4[0] = ((const U4*)q)[0];
                fb.u4[1] = ((const U4*)q)[1];
                o[j] = WMMA_BF16(fa.v, fb.v, o[j]);
            }
        }
    }

    // normalize and store O (bf16, [B*S, E] layout)
#pragma unroll
    for (int j = 0; j < 4; ++j)
#pragma unroll
        for (int r = 0; r < 8; ++r) {
            int qrow = q0 + wave * 16 + lhalf * 8 + r;
            int d    = j * 16 + l15;
            float v  = o[j][r] * (1.0f / lrow[r]);
            Om[(rowbase + qrow) * E + cb + d] = f2bf(v);
        }
}

// ---------------------------------------------------------------------------
extern "C" void kernel_launch(void* const* d_in, const int* in_sizes, int n_in,
                              void* d_out, int out_size, void* d_ws, size_t ws_size,
                              hipStream_t stream) {
    constexpr int B = 4, S = 2048, E = 1024;
    constexpr int M = B * S;                 // 8192
    const float* x  = (const float*)d_in[0];
    const float* Wq = (const float*)d_in[1];
    const float* Wk = (const float*)d_in[2];
    const float* Wv = (const float*)d_in[3];
    const float* Wo = (const float*)d_in[4];

    size_t off = 0;
    auto wsAlloc = [&](size_t bytes) {
        void* p = (char*)d_ws + off;
        off += (bytes + 255) & ~(size_t)255;
        return p;
    };
    u16* xb  = (u16*)wsAlloc((size_t)M * E * 2);
    u16* wqb = (u16*)wsAlloc((size_t)E * E * 2);
    u16* wkb = (u16*)wsAlloc((size_t)E * E * 2);
    u16* wvb = (u16*)wsAlloc((size_t)E * E * 2);
    u16* wob = (u16*)wsAlloc((size_t)E * E * 2);
    u16* Qb  = (u16*)wsAlloc((size_t)M * E * 2);
    u16* Kb  = (u16*)wsAlloc((size_t)M * E * 2);
    u16* Vb  = (u16*)wsAlloc((size_t)M * E * 2);
    u16* Ob  = (u16*)wsAlloc((size_t)M * E * 2);

    const int nx = M * E, nw = E * E;
    mha_cvt_bf16<<<(nx + 255) / 256, 256, 0, stream>>>(x,  xb,  nx);
    mha_cvt_bf16<<<(nw + 255) / 256, 256, 0, stream>>>(Wq, wqb, nw);
    mha_cvt_bf16<<<(nw + 255) / 256, 256, 0, stream>>>(Wk, wkb, nw);
    mha_cvt_bf16<<<(nw + 255) / 256, 256, 0, stream>>>(Wv, wvb, nw);
    mha_cvt_bf16<<<(nw + 255) / 256, 256, 0, stream>>>(Wo, wob, nw);

    dim3 ggrid(M / 128, E / 128);   // (64, 8)
    mha_gemm_wmma<false><<<ggrid, 256, 0, stream>>>(xb, wqb, Qb, M, E, E);
    mha_gemm_wmma<false><<<ggrid, 256, 0, stream>>>(xb, wkb, Kb, M, E, E);
    mha_gemm_wmma<false><<<ggrid, 256, 0, stream>>>(xb, wvb, Vb, M, E, E);

    mha_flash_attn<<<dim3(B * 16, S / 128), 256, 0, stream>>>(Qb, Kb, Vb, Ob);

    mha_gemm_wmma<true><<<ggrid, 256, 0, stream>>>(Ob, wob, (float*)d_out, M, E, E);
}
